// SkeletonEncoder_56126632624716
// MI455X (gfx1250) — matrix operand order repaired and checked by reading, not compiled, not run on hardware
//
#include <hip/hip_runtime.h>
#include <hip/hip_bf16.h>
#include <math.h>

typedef __attribute__((ext_vector_type(16))) _Float16 v16h;
typedef __attribute__((ext_vector_type(8)))  float    v8f;

#define EPSV 1e-5f
#define BNUM 16
#define TN   2048
#define JN   17
#define MROWS (BNUM*TN*JN)   // 557056, divisible by 16 and 256

// ---- input pointer indices (setup_inputs() dict order) ----
enum {
  IN_X=0, IN_ADJ,
  B0_SW, B0_SB, B0_SG, B0_SBB, B0_SM, B0_SV,
  B0_TW, B0_TB, B0_TG, B0_TBB, B0_TM, B0_TV,
  B0_RW, B0_RB, B0_RG, B0_RBB, B0_RM, B0_RV,
  B1_SW, B1_SB, B1_SG, B1_SBB, B1_SM, B1_SV,
  B1_TW, B1_TB, B1_TG, B1_TBB, B1_TM, B1_TV,
  B1_RW, B1_RB, B1_RG, B1_RBB, B1_RM, B1_RV,
  PROJ_W, PROJ_B, NIN
};
struct Ins { const float* p[NIN]; };

// ---- packed f16 weight tiles (32x16 B-tiles, 512 halfs each), offsets in halfs ----
#define WB_W0T  0        // 3 taps * 2 kchunks * 4 ntiles = 24 tiles (12288 halfs)
#define WB_W1S  12288    // 2 * 8 = 16 tiles (8192)
#define WB_W1R  20480    // 16 tiles (8192)
#define WB_W1T  28672    // 3 * 4 * 8 = 96 tiles (49152)
#define WB_PROJ 77824    // 4 * 8 = 32 tiles (16384) -> end 94208 halfs (<256KB)

// ---- fp32 table offsets (floats), stored at ws+256KB ----
#define F_W0S   0     // 64x3 folded
#define F_B0S   192
#define F_W0R   256
#define F_B0R   448
#define F_B0T   512
#define F_B1S   576
#define F_B1R   704
#define F_B1T   832
#define F_PROJB 960
#define F_TOTAL 1088

static __device__ __forceinline__ v8f wmma16(v16h a, v16h b, v8f c) {
  return __builtin_amdgcn_wmma_f32_16x16x32_f16(false, a, false, b, (short)0, c,
                                                false, false);
}
static __device__ __forceinline__ v16h ldg16(const _Float16* p) {
  return *reinterpret_cast<const v16h*>(p);
}

// =====================================================================
// Pack: fold BN (scale=g/sqrt(v+eps), shift=b-m*scale) into weights+bias,
// convert big weights to f16 32x16 B-tiles: tile element (kk,n) at ((kk>>4)*16+n)*16+(kk&15),
// a lane L of the consuming wave reads halfs [L*16, L*16+16).
// =====================================================================
__global__ void k_pack(Ins in, _Float16* __restrict__ hw, float* __restrict__ fw) {
  const int tid = threadIdx.x;
  if (tid < 64) {
    const int o = tid;
    { float sc = in.p[B0_SG][o] * rsqrtf(in.p[B0_SV][o] + EPSV);
      float sh = in.p[B0_SBB][o] - in.p[B0_SM][o]*sc;
      for (int c=0;c<3;++c) fw[F_W0S + o*3 + c] = in.p[B0_SW][o*3+c]*sc;
      fw[F_B0S + o] = in.p[B0_SB][o]*sc + sh; }
    { float sc = in.p[B0_RG][o] * rsqrtf(in.p[B0_RV][o] + EPSV);
      float sh = in.p[B0_RBB][o] - in.p[B0_RM][o]*sc;
      for (int c=0;c<3;++c) fw[F_W0R + o*3 + c] = in.p[B0_RW][o*3+c]*sc;
      fw[F_B0R + o] = in.p[B0_RB][o]*sc + sh; }
    { float sc = in.p[B0_TG][o] * rsqrtf(in.p[B0_TV][o] + EPSV);
      float sh = in.p[B0_TBB][o] - in.p[B0_TM][o]*sc;
      fw[F_B0T + o] = in.p[B0_TB][o]*sc + sh; }
  }
  if (tid < 128) {
    const int o = tid;
    { float sc = in.p[B1_SG][o] * rsqrtf(in.p[B1_SV][o] + EPSV);
      float sh = in.p[B1_SBB][o] - in.p[B1_SM][o]*sc;
      fw[F_B1S + o] = in.p[B1_SB][o]*sc + sh; }
    { float sc = in.p[B1_RG][o] * rsqrtf(in.p[B1_RV][o] + EPSV);
      float sh = in.p[B1_RBB][o] - in.p[B1_RM][o]*sc;
      fw[F_B1R + o] = in.p[B1_RB][o]*sc + sh; }
    { float sc = in.p[B1_TG][o] * rsqrtf(in.p[B1_TV][o] + EPSV);
      float sh = in.p[B1_TBB][o] - in.p[B1_TM][o]*sc;
      fw[F_B1T + o] = in.p[B1_TB][o]*sc + sh; }
    fw[F_PROJB + o] = in.p[PROJ_B][o];
  }
  // W0T: (O=64,C=64,K=3) row-major [o][c][tap], fold t-BN scale
  for (int idx = tid; idx < 12288; idx += 256) {
    int e = idx & 511, tile = idx >> 9;
    int kk = ((e >> 8) << 4) + (e & 15), n = (e >> 4) & 15;
    int nt = tile & 3, kc = (tile >> 2) & 1, tap = tile >> 3;
    int o = nt*16 + n, c = kc*32 + kk;
    float sc = in.p[B0_TG][o] * rsqrtf(in.p[B0_TV][o] + EPSV);
    hw[WB_W0T + idx] = (_Float16)(in.p[B0_TW][(o*64 + c)*3 + tap] * sc);
  }
  // W1S: (128,64)
  for (int idx = tid; idx < 8192; idx += 256) {
    int e = idx & 511, tile = idx >> 9;
    int kk = ((e >> 8) << 4) + (e & 15), n = (e >> 4) & 15;
    int nt = tile & 7, kc = tile >> 3;
    int o = nt*16 + n, c = kc*32 + kk;
    float sc = in.p[B1_SG][o] * rsqrtf(in.p[B1_SV][o] + EPSV);
    hw[WB_W1S + idx] = (_Float16)(in.p[B1_SW][o*64 + c] * sc);
  }
  // W1R: (128,64)
  for (int idx = tid; idx < 8192; idx += 256) {
    int e = idx & 511, tile = idx >> 9;
    int kk = ((e >> 8) << 4) + (e & 15), n = (e >> 4) & 15;
    int nt = tile & 7, kc = tile >> 3;
    int o = nt*16 + n, c = kc*32 + kk;
    float sc = in.p[B1_RG][o] * rsqrtf(in.p[B1_RV][o] + EPSV);
    hw[WB_W1R + idx] = (_Float16)(in.p[B1_RW][o*64 + c] * sc);
  }
  // W1T: (128,128,3)
  for (int idx = tid; idx < 49152; idx += 256) {
    int e = idx & 511, tile = idx >> 9;
    int kk = ((e >> 8) << 4) + (e & 15), n = (e >> 4) & 15;
    int nt = tile & 7, kc = (tile >> 3) & 3, tap = tile >> 5;
    int o = nt*16 + n, c = kc*32 + kk;
    float sc = in.p[B1_TG][o] * rsqrtf(in.p[B1_TV][o] + EPSV);
    hw[WB_W1T + idx] = (_Float16)(in.p[B1_TW][(o*128 + c)*3 + tap] * sc);
  }
  // PROJ: (128,128), no BN
  for (int idx = tid; idx < 16384; idx += 256) {
    int e = idx & 511, tile = idx >> 9;
    int kk = ((e >> 8) << 4) + (e & 15), n = (e >> 4) & 15;
    int nt = tile & 7, kc = tile >> 3;
    int o = nt*16 + n, c = kc*32 + kk;
    hw[WB_PROJ + idx] = (_Float16)(in.p[PROJ_W][o*128 + c]);
  }
}

// =====================================================================
// Block0 (VALU, K=3 is too small for WMMA): adjacency mix + s-conv+BN+relu
// -> y0 (f16, M x 64); raw r-conv+BN -> res0 (f16, M x 64)
// =====================================================================
__global__ __launch_bounds__(256) void k_block0(const float* __restrict__ x,
                                                const float* __restrict__ adj,
                                                const float* __restrict__ fw,
                                                _Float16* __restrict__ y0,
                                                _Float16* __restrict__ res0) {
  __shared__ float sF[F_TOTAL];
  __shared__ float sAdj[289];
  for (int i = threadIdx.x; i < F_TOTAL; i += 256) sF[i] = fw[i];
  if (threadIdx.x < 289) sAdj[threadIdx.x] = adj[threadIdx.x];
  __syncthreads();

  const int m = blockIdx.x*256 + threadIdx.x;
  const unsigned j  = (unsigned)m % JN;
  const unsigned bt = (unsigned)m / JN;
  const unsigned t  = bt & (TN-1);
  const unsigned b  = bt >> 11;

  float xr[3], xa[3];
  #pragma unroll
  for (int c = 0; c < 3; ++c) {
    const float* xp = x + ((size_t)(b*3 + c)*TN + t)*JN;
    xr[c] = xp[j];
    float s = 0.f;
    #pragma unroll
    for (int jj = 0; jj < JN; ++jj) s += xp[jj] * sAdj[jj*JN + j];
    xa[c] = s;
  }
  _Float16* yq = y0   + (size_t)m*64;
  _Float16* rq = res0 + (size_t)m*64;
  for (int o = 0; o < 64; ++o) {
    float as = sF[F_B0S + o], ar = sF[F_B0R + o];
    #pragma unroll
    for (int c = 0; c < 3; ++c) {
      as += sF[F_W0S + o*3 + c] * xa[c];
      ar += sF[F_W0R + o*3 + c] * xr[c];
    }
    yq[o] = (_Float16)fmaxf(as, 0.f);
    rq[o] = (_Float16)ar;
  }
}

// =====================================================================
// tconv0 (WMMA): out0 = relu( BN(tconv3(y0)) + res0 ), 64->64, K=3*64
// One wave = 16 M-rows x 64 N; 3 taps x 2 k-chunks x 4 n-tiles = 24 WMMA
// =====================================================================
__global__ __launch_bounds__(256) void k_tconv0(const _Float16* __restrict__ y0,
                                                const _Float16* __restrict__ res0,
                                                const _Float16* __restrict__ hw,
                                                const float* __restrict__ fw,
                                                _Float16* __restrict__ out0) {
  __shared__ _Float16 sW[12288];
  __shared__ float sB[64];
  for (int i = threadIdx.x; i < 1536; i += 256)
    reinterpret_cast<uint4*>(sW)[i] = reinterpret_cast<const uint4*>(hw + WB_W0T)[i];
  if (threadIdx.x < 64) sB[threadIdx.x] = fw[F_B0T + threadIdx.x];
  __syncthreads();

  const int lane    = threadIdx.x & 31;
  const int tile    = blockIdx.x*8 + (threadIdx.x >> 5);
  const int baseRow = tile << 4;
  const int lrow    = lane & 15;
  const int hi      = lane >> 4;
  const int myRow   = baseRow + lrow;
  const unsigned t  = ((unsigned)myRow / JN) & (TN-1);

  v16h A[3][2] = {};
  #pragma unroll
  for (int tap = 0; tap < 3; ++tap) {
    const bool valid = (tap == 1) || (tap == 0 ? (t > 0u) : (t < TN-1u));
    if (valid) {
      const _Float16* rp = y0 + (size_t)(myRow + (tap-1)*JN)*64 + hi*16;
      A[tap][0] = ldg16(rp);
      A[tap][1] = ldg16(rp + 32);
    }
  }
  v8f acc[4] = {};
  #pragma unroll
  for (int nt = 0; nt < 4; ++nt)
    #pragma unroll
    for (int tap = 0; tap < 3; ++tap)
      #pragma unroll
      for (int ch = 0; ch < 2; ++ch) {
        v16h bm = *reinterpret_cast<const v16h*>(sW + (((tap*2+ch)*4 + nt)*512) + lane*16);
        acc[nt] = wmma16(A[tap][ch], bm, acc[nt]);
      }
  #pragma unroll
  for (int nt = 0; nt < 4; ++nt)
    #pragma unroll
    for (int v = 0; v < 8; ++v) {
      const int row = baseRow + v + (hi << 3);
      const int col = nt*16 + lrow;
      float r = acc[nt][v] + sB[col] + (float)res0[(size_t)row*64 + col];
      out0[(size_t)row*64 + col] = (_Float16)fmaxf(r, 0.f);
    }
}

// =====================================================================
// Adjacency mix on 64 channels: zmix[(bt,k),c] = sum_j out0[(bt,j),c]*adj[j,k]
// =====================================================================
__global__ __launch_bounds__(256) void k_adjmix(const _Float16* __restrict__ out0,
                                                const float* __restrict__ adj,
                                                _Float16* __restrict__ zmix) {
  __shared__ float sAdj[289];
  if (threadIdx.x < 289) sAdj[threadIdx.x] = adj[threadIdx.x];
  __syncthreads();
  const int tid = blockIdx.x*256 + threadIdx.x;
  const int m = tid >> 2, c0 = (tid & 3) * 16;
  const unsigned k = (unsigned)m % JN;
  const size_t base = (size_t)((unsigned)m / JN) * JN;
  float acc[16] = {};
  for (int j = 0; j < JN; ++j) {
    const float w = sAdj[j*JN + k];
    const _Float16* p = out0 + (base + j)*64 + c0;
    #pragma unroll
    for (int i = 0; i < 16; ++i) acc[i] += w * (float)p[i];
  }
  _Float16* q = zmix + (size_t)m*64 + c0;
  #pragma unroll
  for (int i = 0; i < 16; ++i) q[i] = (_Float16)acc[i];
}

// =====================================================================
// Block1 s-conv & r-conv (WMMA, 64->128): y1 = relu(BN(sconv(zmix))),
// res1 = BN(rconv(out0)).  16 + 16 WMMA per wave.
// =====================================================================
__global__ __launch_bounds__(256) void k_conv1(const _Float16* __restrict__ zmix,
                                               const _Float16* __restrict__ out0,
                                               const _Float16* __restrict__ hw,
                                               const float* __restrict__ fw,
                                               _Float16* __restrict__ y1,
                                               _Float16* __restrict__ res1) {
  __shared__ _Float16 sW[16384];                 // W1S then W1R (contiguous in hw)
  __shared__ float sB[256];
  for (int i = threadIdx.x; i < 2048; i += 256)
    reinterpret_cast<uint4*>(sW)[i] = reinterpret_cast<const uint4*>(hw + WB_W1S)[i];
  if (threadIdx.x < 128) {
    sB[threadIdx.x]       = fw[F_B1S + threadIdx.x];
    sB[128 + threadIdx.x] = fw[F_B1R + threadIdx.x];
  }
  __syncthreads();

  const int lane    = threadIdx.x & 31;
  const int tile    = blockIdx.x*8 + (threadIdx.x >> 5);
  const int baseRow = tile << 4;
  const int lrow    = lane & 15;
  const int hi      = lane >> 4;
  const int myRow   = baseRow + lrow;

  const _Float16* zp = zmix + (size_t)myRow*64 + hi*16;
  const _Float16* op = out0 + (size_t)myRow*64 + hi*16;
  v16h As[2] = { ldg16(zp), ldg16(zp + 32) };
  v16h Ar[2] = { ldg16(op), ldg16(op + 32) };

  v8f accS[8] = {};
  #pragma unroll
  for (int nt = 0; nt < 8; ++nt)
    #pragma unroll
    for (int ch = 0; ch < 2; ++ch) {
      v16h bm = *reinterpret_cast<const v16h*>(sW + ((ch*8 + nt)*512) + lane*16);
      accS[nt] = wmma16(As[ch], bm, accS[nt]);
    }
  #pragma unroll
  for (int nt = 0; nt < 8; ++nt)
    #pragma unroll
    for (int v = 0; v < 8; ++v) {
      const int row = baseRow + v + (hi << 3);
      const int col = nt*16 + lrow;
      y1[(size_t)row*128 + col] = (_Float16)fmaxf(accS[nt][v] + sB[col], 0.f);
    }

  v8f accR[8] = {};
  #pragma unroll
  for (int nt = 0; nt < 8; ++nt)
    #pragma unroll
    for (int ch = 0; ch < 2; ++ch) {
      v16h bm = *reinterpret_cast<const v16h*>(sW + 8192 + ((ch*8 + nt)*512) + lane*16);
      accR[nt] = wmma16(Ar[ch], bm, accR[nt]);
    }
  #pragma unroll
  for (int nt = 0; nt < 8; ++nt)
    #pragma unroll
    for (int v = 0; v < 8; ++v) {
      const int row = baseRow + v + (hi << 3);
      const int col = nt*16 + lrow;
      res1[(size_t)row*128 + col] = (_Float16)(accR[nt][v] + sB[128 + col]);
    }
}

// =====================================================================
// tconv1 (WMMA): out1 = relu( BN(tconv3(y1)) + res1 ), 128->128, K=3*128
// 3 taps x 4 k-chunks x 8 n-tiles = 96 WMMA per wave. B from L2 (96KB > LDS budget).
// =====================================================================
__global__ __launch_bounds__(256) void k_tconv1(const _Float16* __restrict__ y1,
                                                const _Float16* __restrict__ res1,
                                                const _Float16* __restrict__ hw,
                                                const float* __restrict__ fw,
                                                _Float16* __restrict__ out1) {
  __shared__ float sB[128];
  if (threadIdx.x < 128) sB[threadIdx.x] = fw[F_B1T + threadIdx.x];
  __syncthreads();

  const int lane    = threadIdx.x & 31;
  const int tile    = blockIdx.x*8 + (threadIdx.x >> 5);
  const int baseRow = tile << 4;
  const int lrow    = lane & 15;
  const int hi      = lane >> 4;
  const int myRow   = baseRow + lrow;
  const unsigned t  = ((unsigned)myRow / JN) & (TN-1);

  __builtin_prefetch(y1 + (size_t)(myRow + JN)*128, 0, 1);

  v8f acc[8] = {};
  #pragma unroll
  for (int tap = 0; tap < 3; ++tap) {
    v16h At[4] = {};
    const bool valid = (tap == 1) || (tap == 0 ? (t > 0u) : (t < TN-1u));
    if (valid) {
      const _Float16* rp = y1 + (size_t)(myRow + (tap-1)*JN)*128 + hi*16;
      At[0] = ldg16(rp);       At[1] = ldg16(rp + 32);
      At[2] = ldg16(rp + 64);  At[3] = ldg16(rp + 96);
    }
    const _Float16* wb = hw + WB_W1T + (size_t)tap*32*512 + lane*16;
    #pragma unroll
    for (int ch = 0; ch < 4; ++ch)
      #pragma unroll
      for (int nt = 0; nt < 8; ++nt) {
        v16h bm = *reinterpret_cast<const v16h*>(wb + (ch*8 + nt)*512);
        acc[nt] = wmma16(At[ch], bm, acc[nt]);
      }
  }
  #pragma unroll
  for (int nt = 0; nt < 8; ++nt)
    #pragma unroll
    for (int v = 0; v < 8; ++v) {
      const int row = baseRow + v + (hi << 3);
      const int col = nt*16 + lrow;
      float r = acc[nt][v] + sB[col] + (float)res1[(size_t)row*128 + col];
      out1[(size_t)row*128 + col] = (_Float16)fmaxf(r, 0.f);
    }
}

// =====================================================================
// Mean over joints: meanY[bt,c] = (1/17) * sum_j out1[(bt,j),c]
// =====================================================================
__global__ __launch_bounds__(256) void k_mean(const _Float16* __restrict__ out1,
                                              _Float16* __restrict__ meanY) {
  const int tid = blockIdx.x*256 + threadIdx.x;
  const int bt = tid >> 3, c0 = (tid & 7) * 16;
  float acc[16] = {};
  const _Float16* base = out1 + (size_t)bt*JN*128 + c0;
  for (int j = 0; j < JN; ++j) {
    const _Float16* p = base + (size_t)j*128;
    #pragma unroll
    for (int i = 0; i < 16; ++i) acc[i] += (float)p[i];
  }
  const float inv = 1.0f / 17.0f;
  _Float16* q = meanY + (size_t)bt*128 + c0;
  #pragma unroll
  for (int i = 0; i < 16; ++i) q[i] = (_Float16)(acc[i] * inv);
}

// =====================================================================
// Projection (WMMA, 128->128, M=B*T): out[bt,e] = meanY[bt,:]·proj_w[e,:] + pb[e]
// =====================================================================
__global__ __launch_bounds__(256) void k_proj(const _Float16* __restrict__ meanY,
                                              const _Float16* __restrict__ hw,
                                              const float* __restrict__ fw,
                                              float* __restrict__ out) {
  __shared__ _Float16 sW[16384];
  __shared__ float sB[128];
  for (int i = threadIdx.x; i < 2048; i += 256)
    reinterpret_cast<uint4*>(sW)[i] = reinterpret_cast<const uint4*>(hw + WB_PROJ)[i];
  if (threadIdx.x < 128) sB[threadIdx.x] = fw[F_PROJB + threadIdx.x];
  __syncthreads();

  const int lane    = threadIdx.x & 31;
  const int tile    = blockIdx.x*8 + (threadIdx.x >> 5);
  const int baseRow = tile << 4;
  const int lrow    = lane & 15;
  const int hi      = lane >> 4;
  const int myRow   = baseRow + lrow;

  const _Float16* rp = meanY + (size_t)myRow*128 + hi*16;
  v16h A[4] = { ldg16(rp), ldg16(rp + 32), ldg16(rp + 64), ldg16(rp + 96) };

  v8f acc[8] = {};
  #pragma unroll
  for (int nt = 0; nt < 8; ++nt)
    #pragma unroll
    for (int ch = 0; ch < 4; ++ch) {
      v16h bm = *reinterpret_cast<const v16h*>(sW + ((ch*8 + nt)*512) + lane*16);
      acc[nt] = wmma16(A[ch], bm, acc[nt]);
    }
  #pragma unroll
  for (int nt = 0; nt < 8; ++nt)
    #pragma unroll
    for (int v = 0; v < 8; ++v) {
      const int row = baseRow + v + (hi << 3);
      const int col = nt*16 + lrow;
      out[(size_t)row*128 + col] = acc[nt][v] + sB[col];
    }
}

// =====================================================================
extern "C" void kernel_launch(void* const* d_in, const int* in_sizes, int n_in,
                              void* d_out, int out_size, void* d_ws, size_t ws_size,
                              hipStream_t stream) {
  (void)in_sizes; (void)n_in; (void)out_size; (void)ws_size;
  Ins ins;
  for (int i = 0; i < NIN; ++i) ins.p[i] = (const float*)d_in[i];

  _Float16* hw  = (_Float16*)d_ws;                          // packed f16 weights
  float*    fw  = (float*)((char*)d_ws + (256u << 10));     // folded fp32 table
  _Float16* act = (_Float16*)((char*)d_ws + (1u << 20));    // activation arena

  const size_t SZ64 = (size_t)MROWS * 64;   // halfs of one 64-ch tensor
  _Float16* y0    = act;                    // [0, 1)·SZ64
  _Float16* res0  = act + SZ64;             // [1, 2)
  _Float16* out0  = act + 2*SZ64;           // [2, 3)
  _Float16* zmix  = act;                    // reuse y0 region (y0 dead)
  _Float16* y1    = act + 3*SZ64;           // [3, 5)
  _Float16* res1  = act + 5*SZ64;           // [5, 7)
  _Float16* out1  = act;                    // reuse [0, 2) (y0/res0/zmix dead)
  _Float16* meanY = act + 2*SZ64;           // reuse out0 region (dead)
  float* out = (float*)d_out;

  k_pack  <<<1,            256, 0, stream>>>(ins, hw, fw);
  k_block0<<<MROWS/256,    256, 0, stream>>>(ins.p[IN_X], ins.p[IN_ADJ], fw, y0, res0);
  k_tconv0<<<MROWS/16/8,   256, 0, stream>>>(y0, res0, hw, fw, out0);
  k_adjmix<<<MROWS*4/256,  256, 0, stream>>>(out0, ins.p[IN_ADJ], zmix);
  k_conv1 <<<MROWS/16/8,   256, 0, stream>>>(zmix, out0, hw, fw, y1, res1);
  k_tconv1<<<MROWS/16/8,   256, 0, stream>>>(y1, res1, hw, fw, out1);
  k_mean  <<<(BNUM*TN*8)/256, 256, 0, stream>>>(out1, meanY);
  k_proj  <<<(BNUM*TN/16)/8,  256, 0, stream>>>(meanY, hw, fw, out);
}
